// Pnet_12764642804109
// MI455X (gfx1250) — matrix operand order repaired and checked by reading, not compiled, not run on hardware
//
#include <hip/hip_runtime.h>

// ---------------------------------------------------------------------------
// MI455X (gfx1250):  out = mu + softplus(sig)*noise over GRU pipeline.
// All GEMMs via v_wmma_f32_16x16x32_bf16. Weights prepacked to B-fragment
// layout (2x b128 per fragment). GRU: 8 persistent WGs (batch-parallel),
// h double-buffered in LDS, xp tile prefetched with async global->LDS
// (ASYNCcnt). GRU k-loop shares one A fragment across 6 independent WMMA
// chains (3 gates x 2 column tiles) to cover the WMMA->WMMA RAW hazard.
// ---------------------------------------------------------------------------

#define Bsz 128
#define Wsz 1024
#define Zsz 128
#define Hsz 512
#define G3  1536
#define Dsz 512
#define Osz 128

typedef __attribute__((ext_vector_type(16))) __bf16 bf16x16;
typedef __attribute__((ext_vector_type(8)))  float  f32x8;
typedef int v4i __attribute__((vector_size(16)));

struct Frag { union { unsigned int u[8]; bf16x16 v; }; };

#if defined(__has_builtin)
#  if __has_builtin(__builtin_amdgcn_global_load_async_to_lds_b128)
#    define ASYNC_LDS 1
#  endif
#endif

__device__ __forceinline__ unsigned short f2bf(float f) {
    union { float f; unsigned int u; } x; x.f = f;
    unsigned int u = x.u;
    u += 0x7FFFu + ((u >> 16) & 1u);   // round-to-nearest-even
    return (unsigned short)(u >> 16);
}
__device__ __forceinline__ float bf2f(unsigned short h) {
    union { unsigned int u; float f; } x; x.u = ((unsigned int)h) << 16; return x.f;
}
__device__ __forceinline__ unsigned int pack2bf(float a, float b) {
    return (unsigned int)f2bf(a) | ((unsigned int)f2bf(b) << 16);
}

// 16-byte global -> LDS copy; async (ASYNCcnt) when available.
__device__ __forceinline__ void copy16_g2l(const void* g, void* l) {
#ifdef ASYNC_LDS
    __builtin_amdgcn_global_load_async_to_lds_b128(
        (__attribute__((address_space(1))) v4i*)g,
        (__attribute__((address_space(3))) v4i*)l, 0, 0);
#else
    *(uint4*)l = *(const uint4*)g;
#endif
}
__device__ __forceinline__ void async_wait_all() {
#ifdef ASYNC_LDS
#  if __has_builtin(__builtin_amdgcn_s_wait_asynccnt)
    __builtin_amdgcn_s_wait_asynccnt(0);
#  else
    asm volatile("s_wait_asynccnt 0x0" ::: "memory");
#  endif
#endif
}

__device__ __forceinline__ f32x8 wmma_bf16(bf16x16 a, bf16x16 b, f32x8 c) {
    return __builtin_amdgcn_wmma_f32_16x16x32_bf16(
        false, a, false, b, (short)0, c, false, false);
}

// A fragment (16x32 bf16) from LDS, row-major (row_stride in dwords).
__device__ __forceinline__ bf16x16 load_afrag(const unsigned int* base,
                                              int row_stride_dw, int kt, int lane) {
    int m  = lane & 15;
    int kb = (lane >> 4) << 3;
    const unsigned int* row = base + m * row_stride_dw;
    int d0 = ((kt << 5) + kb) >> 1;
    Frag f;
#pragma unroll
    for (int i = 0; i < 4; ++i) f.u[i]     = row[d0 + i];
#pragma unroll
    for (int i = 0; i < 4; ++i) f.u[4 + i] = row[d0 + 8 + i];
    return f.v;
}

// B fragment (32x16 bf16) from prepacked weights: 2x b128 per lane.
__device__ __forceinline__ bf16x16 load_bfrag(const unsigned int* Bp,
                                              int nt, int kt, int ktiles, int lane) {
    const unsigned int* p = Bp + ((((size_t)nt * ktiles + kt) * 32 + lane) << 3);
    uint4 lo = ((const uint4*)p)[0];
    uint4 hi = ((const uint4*)p)[1];
    Frag f;
    f.u[0] = lo.x; f.u[1] = lo.y; f.u[2] = lo.z; f.u[3] = lo.w;
    f.u[4] = hi.x; f.u[5] = hi.y; f.u[6] = hi.z; f.u[7] = hi.w;
    return f.v;
}

// ---------------------------------------------------------------------------
// Pack W (N x K row-major f32) into bf16 B-fragment tiles.
// ---------------------------------------------------------------------------
__global__ void pack_b_kernel(const float* __restrict__ W,
                              unsigned int* __restrict__ out, int N, int K) {
    int idx = blockIdx.x * blockDim.x + threadIdx.x;
    int ktiles = K >> 5;
    int total = (N >> 4) * ktiles * 256;
    if (idx >= total) return;
    int v    = idx & 7;
    int lane = (idx >> 3) & 31;
    int tile = idx >> 8;
    int kt = tile % ktiles;
    int nt = tile / ktiles;
    int n = (nt << 4) + (lane & 15);
    int k = (kt << 5) + ((lane >> 4) << 3) + ((v & 3) << 1) + ((v >> 2) << 4);
    const float* wr = W + (size_t)n * K + k;
    out[idx] = pack2bf(wr[0], wr[1]);
}

// ---------------------------------------------------------------------------
// Fused zt = z@Bl+bl, xp = zt@Bih+bih; xp stored bf16, time-major r = t*B+b.
// ---------------------------------------------------------------------------
__global__ __launch_bounds__(256) void zt_xp_kernel(
    const float* __restrict__ z, const unsigned int* __restrict__ Bl,
    const float* __restrict__ bl, const unsigned int* __restrict__ Bih,
    const float* __restrict__ bih, unsigned short* __restrict__ xpb) {
    __shared__ __align__(16) unsigned int az[16][64];
    __shared__ __align__(16) float        zf[16][128];
    __shared__ __align__(16) unsigned int azt[16][64];

    int mt = blockIdx.x;
    int t  = mt >> 3;
    int b0 = (mt & 7) << 4;
    int tid = threadIdx.x, lane = tid & 31, wave = tid >> 5;

    { // stage z tile -> bf16 LDS
        int row = tid >> 4, pj = tid & 15;
        const float* zr = z + ((size_t)(b0 + row) * Wsz + t) * Zsz;
#pragma unroll
        for (int i = 0; i < 4; ++i) {
            int p = pj * 4 + i;
            az[row][p] = pack2bf(zr[2 * p], zr[2 * p + 1]);
        }
    }
    __syncthreads();

    { // zt
        int nt = wave;
        f32x8 acc = {};
#pragma unroll
        for (int kt = 0; kt < 4; ++kt) {
            bf16x16 a = load_afrag(&az[0][0], 64, kt, lane);
            acc = wmma_bf16(a, load_bfrag(Bl, nt, kt, 4, lane), acc);
        }
        int n = (nt << 4) + (lane & 15);
        float bias = bl[n];
        int mb = (lane >> 4) << 3;
#pragma unroll
        for (int v = 0; v < 8; ++v) zf[mb + v][n] = acc[v] + bias;
    }
    __syncthreads();

    { // repack zt -> bf16
        int row = tid >> 4, pj = tid & 15;
#pragma unroll
        for (int i = 0; i < 4; ++i) {
            int p = pj * 4 + i;
            azt[row][p] = pack2bf(zf[row][2 * p], zf[row][2 * p + 1]);
        }
    }
    __syncthreads();

    for (int j = 0; j < 12; ++j) {   // xp: 96 n-tiles, 12 per wave
        int nt = wave * 12 + j;
        f32x8 acc = {};
#pragma unroll
        for (int kt = 0; kt < 4; ++kt) {
            bf16x16 a = load_afrag(&azt[0][0], 64, kt, lane);
            acc = wmma_bf16(a, load_bfrag(Bih, nt, kt, 4, lane), acc);
        }
        int n = (nt << 4) + (lane & 15);
        float bias = bih[n];
        int mb = (lane >> 4) << 3;
        size_t rbase = (size_t)mt << 4;
#pragma unroll
        for (int v = 0; v < 8; ++v)
            xpb[(rbase + mb + v) * G3 + n] = f2bf(acc[v] + bias);
    }
}

// ---------------------------------------------------------------------------
// Persistent GRU: 8 blocks x 16 batch rows. h double-buffered bf16 in LDS;
// xp step-tile (48KB, contiguous) double-buffered in LDS via async prefetch.
// Per k-tile: 1 shared A fragment -> 6 independent WMMAs (3 gates x 2 tiles).
// ---------------------------------------------------------------------------
__global__ __launch_bounds__(512) void gru_kernel(
    const unsigned short* __restrict__ xpb, const unsigned int* __restrict__ Bhh,
    const float* __restrict__ bhh, unsigned short* __restrict__ hs) {
    __shared__ __align__(16) unsigned short hb[2][16][Hsz];   //  32 KB
    __shared__ __align__(16) unsigned short xq[2][16][G3];    //  96 KB

    int tid = threadIdx.x, lane = tid & 31, wave = tid >> 5;
    int b0 = blockIdx.x << 4;

    for (int i = tid; i < 16 * Hsz; i += 512)
        ((unsigned short*)&hb[0][0][0])[i] = 0;

    // prefetch xp tile for t=0 (16 rows x 1536 bf16 = 3072 x 16B, contiguous)
    {
        const unsigned short* src = xpb + (size_t)b0 * G3;
        char* dst = (char*)&xq[0][0][0];
        for (int i = 0; i < 6; ++i) {
            int q = tid + i * 512;
            copy16_g2l(src + q * 8, dst + q * 16);
        }
    }
    async_wait_all();
    __syncthreads();

    for (int t = 0; t < Wsz; ++t) {
        int cur = t & 1, nxt = cur ^ 1;

        { // async prefetch next step's xp tile into the other buffer
            int tt = (t + 1 < Wsz) ? t + 1 : Wsz - 1;
            const unsigned short* src = xpb + ((size_t)tt * Bsz + b0) * G3;
            char* dst = (char*)&xq[nxt][0][0];
            for (int i = 0; i < 6; ++i) {
                int q = tid + i * 512;
                copy16_g2l(src + q * 8, dst + q * 16);
            }
        }

        const unsigned int* hcur = (const unsigned int*)&hb[cur][0][0];
        int ct0 = wave * 2, ct1 = ct0 + 1;    // this wave's two column tiles
        f32x8 acc[6] = {};                    // [gate][tile]: r0,r1,u0,u1,n0,n1
        for (int kt = 0; kt < 16; ++kt) {     // K = H = 512
            bf16x16 a = load_afrag(hcur, Hsz / 2, kt, lane);
            acc[0] = wmma_bf16(a, load_bfrag(Bhh,      ct0, kt, 16, lane), acc[0]);
            acc[1] = wmma_bf16(a, load_bfrag(Bhh,      ct1, kt, 16, lane), acc[1]);
            acc[2] = wmma_bf16(a, load_bfrag(Bhh, 32 + ct0, kt, 16, lane), acc[2]);
            acc[3] = wmma_bf16(a, load_bfrag(Bhh, 32 + ct1, kt, 16, lane), acc[3]);
            acc[4] = wmma_bf16(a, load_bfrag(Bhh, 64 + ct0, kt, 16, lane), acc[4]);
            acc[5] = wmma_bf16(a, load_bfrag(Bhh, 64 + ct1, kt, 16, lane), acc[5]);
        }
#pragma unroll
        for (int j = 0; j < 2; ++j) {
            int ct = ct0 + j;
            int n = (ct << 4) + (lane & 15);
            float br = bhh[n], bu = bhh[Hsz + n], bn = bhh[2 * Hsz + n];
            int mb = (lane >> 4) << 3;
#pragma unroll
            for (int v = 0; v < 8; ++v) {
                int m = mb + v;
                float xr = bf2f(xq[cur][m][n]);
                float xu = bf2f(xq[cur][m][Hsz + n]);
                float xn = bf2f(xq[cur][m][2 * Hsz + n]);
                float rg = 1.f / (1.f + __expf(-(xr + acc[0 + j][v] + br)));
                float ug = 1.f / (1.f + __expf(-(xu + acc[2 + j][v] + bu)));
                float ng = tanhf(xn + rg * (acc[4 + j][v] + bn));
                float hp = bf2f(hb[cur][m][n]);
                float hnew = (1.f - ug) * ng + ug * hp;
                unsigned short h16 = f2bf(hnew);
                hb[nxt][m][n] = h16;
                hs[((size_t)(b0 + m) * Wsz + t) * Hsz + n] = h16;
            }
        }
        async_wait_all();
        __syncthreads();
    }
}

// ---------------------------------------------------------------------------
// dense = hs @ Bd + bd. 64-row M tiles: each B fragment reused by 4 A tiles.
// ---------------------------------------------------------------------------
__global__ __launch_bounds__(256) void dense_kernel(
    const unsigned short* __restrict__ hsm, const unsigned int* __restrict__ Bd,
    const float* __restrict__ bd, unsigned short* __restrict__ dn) {
    __shared__ __align__(16) unsigned int ah[64][256];   // 64 x 512 bf16 = 64 KB
    int mt = blockIdx.x, tid = threadIdx.x, lane = tid & 31, wave = tid >> 5;
    size_t r0 = (size_t)mt << 6;

    {
        int row = tid >> 2, seg = tid & 3;   // 4 threads/row, 64 dwords each
        const unsigned short* src = hsm + (r0 + row) * Dsz + seg * 128;
        for (int i = 0; i < 16; ++i)
            copy16_g2l(src + i * 8, &ah[row][seg * 64 + i * 4]);
    }
    async_wait_all();
    __syncthreads();

    for (int j = 0; j < 4; ++j) {
        int nt = wave * 4 + j;
        f32x8 acc[4] = {};
        for (int kt = 0; kt < 16; ++kt) {
            bf16x16 bm = load_bfrag(Bd, nt, kt, 16, lane);
#pragma unroll
            for (int ms = 0; ms < 4; ++ms) {
                bf16x16 a = load_afrag(&ah[ms * 16][0], 256, kt, lane);
                acc[ms] = wmma_bf16(a, bm, acc[ms]);
            }
        }
        int n = (nt << 4) + (lane & 15);
        float bias = bd[n];
        int mb = (lane >> 4) << 3;
#pragma unroll
        for (int ms = 0; ms < 4; ++ms)
#pragma unroll
            for (int v = 0; v < 8; ++v)
                dn[(r0 + ms * 16 + mb + v) * Dsz + n] = f2bf(acc[ms][v] + bias);
    }
}

// ---------------------------------------------------------------------------
// Head: mu & sigma GEMMs fused with softplus + noise. 32-row M tiles.
// ---------------------------------------------------------------------------
__global__ __launch_bounds__(256) void head_kernel(
    const unsigned short* __restrict__ dnm,
    const unsigned int* __restrict__ Bmu, const float* __restrict__ bmu,
    const unsigned int* __restrict__ Bsg, const float* __restrict__ bsg,
    const float* __restrict__ noise, float* __restrict__ out) {
    __shared__ __align__(16) unsigned int ad[32][256];   // 32 KB
    int mt = blockIdx.x, tid = threadIdx.x, lane = tid & 31, wave = tid >> 5;
    size_t r0 = (size_t)mt << 5;

    {
        int row = tid >> 3, seg = tid & 7;   // 8 threads/row, 32 dwords each
        const unsigned short* src = dnm + (r0 + row) * Dsz + seg * 64;
        for (int i = 0; i < 8; ++i)
            copy16_g2l(src + i * 8, &ad[row][seg * 32 + i * 4]);
    }
    async_wait_all();
    __syncthreads();

    int nt = wave;                           // O=128 -> 8 n-tiles per gate
    f32x8 am[2] = {}, as[2] = {};
    for (int kt = 0; kt < 16; ++kt) {        // K = D = 512
        bf16x16 bm = load_bfrag(Bmu, nt, kt, 16, lane);
        bf16x16 bs = load_bfrag(Bsg, nt, kt, 16, lane);
#pragma unroll
        for (int ms = 0; ms < 2; ++ms) {
            bf16x16 a = load_afrag(&ad[ms * 16][0], 256, kt, lane);
            am[ms] = wmma_bf16(a, bm, am[ms]);
            as[ms] = wmma_bf16(a, bs, as[ms]);
        }
    }
    int n = (nt << 4) + (lane & 15);
    float bm_ = bmu[n], bs_ = bsg[n];
    int mb = (lane >> 4) << 3;
#pragma unroll
    for (int ms = 0; ms < 2; ++ms)
#pragma unroll
        for (int v = 0; v < 8; ++v) {
            size_t r = r0 + ms * 16 + mb + v;
            float mu = am[ms][v] + bm_;
            float sv = as[ms][v] + bs_;
            float sp = fmaxf(sv, 0.f) + log1pf(__expf(-fabsf(sv)));
            out[r * Osz + n] = mu + sp * noise[r * Osz + n];
        }
}

// ---------------------------------------------------------------------------
extern "C" void kernel_launch(void* const* d_in, const int* in_sizes, int n_in,
                              void* d_out, int out_size, void* d_ws, size_t ws_size,
                              hipStream_t stream) {
    (void)in_sizes; (void)n_in; (void)out_size; (void)ws_size;
    const float* z     = (const float*)d_in[0];
    const float* noise = (const float*)d_in[1];
    const float* W_l   = (const float*)d_in[2];
    const float* b_l   = (const float*)d_in[3];
    const float* W_ih  = (const float*)d_in[4];
    const float* b_ih  = (const float*)d_in[5];
    const float* W_hh  = (const float*)d_in[6];
    const float* b_hh  = (const float*)d_in[7];
    const float* W_d   = (const float*)d_in[8];
    const float* b_d   = (const float*)d_in[9];
    const float* W_mu  = (const float*)d_in[10];
    const float* b_mu  = (const float*)d_in[11];
    const float* W_sg  = (const float*)d_in[12];
    const float* b_sg  = (const float*)d_in[13];
    float* out = (float*)d_out;

    char* ws = (char*)d_ws;                       // 256B-aligned offsets
    unsigned int*   pWl  = (unsigned int*)(ws + 0);
    unsigned int*   pWih = (unsigned int*)(ws + 32768);
    unsigned int*   pWhh = (unsigned int*)(ws + 425984);
    unsigned int*   pWd  = (unsigned int*)(ws + 1998848);
    unsigned int*   pWmu = (unsigned int*)(ws + 2523136);
    unsigned int*   pWsg = (unsigned int*)(ws + 2654208);
    unsigned short* xpb  = (unsigned short*)(ws + 2785280);            // 384 MB bf16
    unsigned short* hs   = (unsigned short*)(ws + 2785280 + 402653184ull);  // 128 MB
    unsigned short* dn   = (unsigned short*)(ws + 2785280 + 402653184ull
                                                + 134217728ull);            // 128 MB

    pack_b_kernel<<<32,   256, 0, stream>>>(W_l,  pWl,  Zsz, Zsz);
    pack_b_kernel<<<384,  256, 0, stream>>>(W_ih, pWih, G3,  Zsz);
    pack_b_kernel<<<1536, 256, 0, stream>>>(W_hh, pWhh, G3,  Hsz);
    pack_b_kernel<<<512,  256, 0, stream>>>(W_d,  pWd,  Dsz, Hsz);
    pack_b_kernel<<<128,  256, 0, stream>>>(W_mu, pWmu, Osz, Dsz);
    pack_b_kernel<<<128,  256, 0, stream>>>(W_sg, pWsg, Osz, Dsz);

    zt_xp_kernel<<<(Bsz * Wsz) / 16, 256, 0, stream>>>(z, pWl, b_l, pWih, b_ih, xpb);
    gru_kernel  <<<Bsz / 16,         512, 0, stream>>>(xpb, pWhh, b_hh, hs);
    dense_kernel<<<(Bsz * Wsz) / 64, 256, 0, stream>>>(hs, pWd, b_d, dn);
    head_kernel <<<(Bsz * Wsz) / 32, 256, 0, stream>>>(dn, pWmu, b_mu, pWsg, b_sg,
                                                       noise, out);
}
